// GraphConvolution_75557064672009
// MI455X (gfx1250) — compile-verified, matrix operand order checked by reading, not run on hardware
//
#include <hip/hip_runtime.h>

typedef __attribute__((ext_vector_type(16))) _Float16 v16h;
typedef __attribute__((ext_vector_type(8)))  _Float16 v8h;
typedef __attribute__((ext_vector_type(8)))  float    v8f;

#define KNBR   7      // neighbors per row
#define FDIM   64     // features per row
#define UNITSN 128    // output units
#define KFLAT  448    // KNBR * FDIM
#define UHALF  64     // units handled per block (grid.y splits 128 -> 2x64)
#define PADK   456    // padded K stride (f16 elems): 228 dwords/row -> conflict-free b128 reads
#define KSTEPS 14     // KFLAT / 32
#define WAVES  8      // waves per block (wave32) -> 128 rows per block

__global__ __launch_bounds__(WAVES * 32, 1) void graphconv_wmma_f16(
    const float* __restrict__ X, const int* __restrict__ G,
    const float* __restrict__ W, const float* __restrict__ bias,
    float* __restrict__ out, int N)
{
    // W half-tile, transposed, f16: 64 * 456 * 2 B = 57 KB LDS
    __shared__ _Float16 Wt[UHALF * PADK];

    const int tid = threadIdx.x;
    const int ub  = blockIdx.y * UHALF;

    // Stage W[:, ub:ub+64] -> LDS transposed (f32 -> f16).
    // Two k-adjacent values packed into one 32-bit DS store; global reads coalesced over u.
    for (int idx = tid; idx < (KFLAT / 2) * UHALF; idx += WAVES * 32) {
        const int u = idx & 63;          // 0..63
        const int k = (idx >> 6) << 1;   // even k: 0,2,..,446
        const float w0 = W[(k + 0) * UNITSN + ub + u];
        const float w1 = W[(k + 1) * UNITSN + ub + u];
        const unsigned short h0 = __builtin_bit_cast(unsigned short, (_Float16)w0);
        const unsigned short h1 = __builtin_bit_cast(unsigned short, (_Float16)w1);
        *(unsigned int*)&Wt[u * PADK + k] = (unsigned int)h0 | ((unsigned int)h1 << 16);
    }
    __syncthreads();

    const int wave = tid >> 5;      // 0..7
    const int lane = tid & 31;      // wave32
    const int lr   = lane & 15;     // row/col within tile
    const int h    = lane >> 4;     // lane half: K-group selector

    const int row0 = (blockIdx.x * WAVES + wave) * 16;   // M-tile base row
    int m = row0 + lr;
    if (m > N - 1) m = N - 1;       // clamp gathers for tail (WMMA needs full EXEC)

    int nbr[KNBR];
#pragma unroll
    for (int j = 0; j < KNBR; ++j) nbr[j] = G[m * KNBR + j];

    v8f acc[4] = { v8f{}, v8f{}, v8f{}, v8f{} };

#pragma unroll
    for (int t = 0; t < KSTEPS; ++t) {
        const int j  = t >> 1;                       // neighbor index
        const int fb = ((t & 1) << 5) + (h << 3);    // feature base for this lane half

        // A fragment: row m of gathered features, K = kb..kb+7 and kb+16..kb+23
        const float* src = X + (long)nbr[j] * FDIM + fb;
        const float4 x0 = *(const float4*)(src +  0);
        const float4 x1 = *(const float4*)(src +  4);
        const float4 x2 = *(const float4*)(src + 16);
        const float4 x3 = *(const float4*)(src + 20);

        v16h a;
        a[0]  = (_Float16)x0.x;  a[1]  = (_Float16)x0.y;
        a[2]  = (_Float16)x0.z;  a[3]  = (_Float16)x0.w;
        a[4]  = (_Float16)x1.x;  a[5]  = (_Float16)x1.y;
        a[6]  = (_Float16)x1.z;  a[7]  = (_Float16)x1.w;
        a[8]  = (_Float16)x2.x;  a[9]  = (_Float16)x2.y;
        a[10] = (_Float16)x2.z;  a[11] = (_Float16)x2.w;
        a[12] = (_Float16)x3.x;  a[13] = (_Float16)x3.y;
        a[14] = (_Float16)x3.z;  a[15] = (_Float16)x3.w;

        const int kb = (t << 5) + (h << 3);
#pragma unroll
        for (int nt = 0; nt < 4; ++nt) {
            const int u = (nt << 4) + lr;
            // two 16B-aligned ds_load_b128 per fragment
            const v8h blo = *(const v8h*)&Wt[u * PADK + kb];
            const v8h bhi = *(const v8h*)&Wt[u * PADK + kb + 16];
            const v16h bf = __builtin_shufflevector(blo, bhi,
                0,1,2,3,4,5,6,7,8,9,10,11,12,13,14,15);
            acc[nt] = __builtin_amdgcn_wmma_f32_16x16x32_f16(
                /*neg_a=*/false, a, /*neg_b=*/false, bf,
                /*c_mod=*/(short)0, acc[nt],
                /*reuse_a=*/false, /*reuse_b=*/false);
        }
    }

    // Epilogue: D layout — VGPR r holds row (row0 + 8*h + r), lane%16 holds column.
    const int orow = row0 + (h << 3);
#pragma unroll
    for (int nt = 0; nt < 4; ++nt) {
        const int u  = ub + (nt << 4) + lr;
        const float bv = bias[u];
        float* op = out + (long)orow * UNITSN + u;
#pragma unroll
        for (int r = 0; r < 8; ++r) {
            if (orow + r < N) op[(long)r * UNITSN] = acc[nt][r] + bv;
        }
    }
}

extern "C" void kernel_launch(void* const* d_in, const int* in_sizes, int n_in,
                              void* d_out, int out_size, void* d_ws, size_t ws_size,
                              hipStream_t stream)
{
    const float* X   = (const float*)d_in[0];   // [N, 64] f32
    const int*   G   = (const int*)  d_in[1];   // [N, 7]  i32
    const float* W   = (const float*)d_in[2];   // [448, 128] f32
    const float* b   = (const float*)d_in[3];   // [128] f32
    float*       out = (float*)d_out;           // [N, 128] f32

    const int N = in_sizes[0] / FDIM;           // 200000
    const int mtiles = (N + 15) / 16;           // 12500
    dim3 grid((mtiles + WAVES - 1) / WAVES, 2); // 1563 x 2
    dim3 block(WAVES * 32);                     // 256
    graphconv_wmma_f16<<<grid, block, 0, stream>>>(X, G, W, b, out, N);
}
